// Tri_Att_GCL_17772574671126
// MI455X (gfx1250) — compile-verified, matrix operand order checked by reading, not run on hardware
//
#include <hip/hip_runtime.h>

// ---------------------------------------------------------------------------
// Triangle attention GCL for gfx1250 (MI455X), wave32 + WMMA bf16.
// Pipeline:
//   0) init   : clear softmax max/denom/sum accumulators
//   0b) zcvt  : Z (f32) -> Zb (bf16) streaming convert
//   0c) packB : weights f32 -> bf16 pre-packed into WMMA B-operand layout
//   1) proj   : q,k,v,g = Z@W[h] via WMMA (b128 operand loads), b = Z@Wb,
//               sigmoid gate folded into g store
//   2) logits : per-(tri,head) dot(q,k)/8 + b[jk]; atomicMax segment max
//   3) accum  : a=exp(l-m); atomicAdd denom, atomicAdd S += a*v[ik]
//   4) final  : val = S/denom (1 if count==0); feat = g*val (bf16)
//   5) outgemm: out = feat(E,256) @ Wout(256,128) + bout via WMMA
// ---------------------------------------------------------------------------

typedef __attribute__((ext_vector_type(16))) __bf16 v16bf;
typedef __attribute__((ext_vector_type(8)))  float  v8f;

#define INV_SQRT_H 0.125f

// float -> bf16 bits (round to nearest even)
__device__ __forceinline__ unsigned short f2bfu(float f) {
    unsigned u = __float_as_uint(f);
    u += 0x7fffu + ((u >> 16) & 1u);
    return (unsigned short)(u >> 16);
}
__device__ __forceinline__ float bfu2f(unsigned short s) {
    return __uint_as_float(((unsigned)s) << 16);
}
__device__ __forceinline__ float lo16(unsigned u) { return __uint_as_float(u << 16); }
__device__ __forceinline__ float hi16(unsigned u) { return __uint_as_float(u & 0xffff0000u); }

// two 16-byte chunks -> one v16bf WMMA operand
struct U8x32 { uint4 a, b; };
__device__ __forceinline__ v16bf load_op(const unsigned short* p0,
                                         const unsigned short* p1) {
    U8x32 t;
    t.a = *(const uint4*)p0;
    t.b = *(const uint4*)p1;
    return __builtin_bit_cast(v16bf, t);
}

// monotonic float<->uint encoding for atomicMax on floats
__device__ __forceinline__ unsigned fenc(float f) {
    unsigned u = __float_as_uint(f);
    return (u & 0x80000000u) ? ~u : (u | 0x80000000u);
}
__device__ __forceinline__ float fdec(unsigned u) {
    return __uint_as_float((u & 0x80000000u) ? (u & 0x7fffffffu) : ~u);
}
#define NEG_INF_ENC 0x007FFFFFu  // fenc(-inf)

// ---------------------------------------------------------------------------
// 0) init accumulators
// ---------------------------------------------------------------------------
__global__ void init_kernel(unsigned* menc, float* denom, float* S,
                            long long nSmall, long long nS) {
    long long i = (long long)blockIdx.x * blockDim.x + threadIdx.x;
    long long stride = (long long)gridDim.x * blockDim.x;
    for (long long j = i; j < nS; j += stride) S[j] = 0.0f;
    for (long long j = i; j < nSmall; j += stride) {
        menc[j] = NEG_INF_ENC;
        denom[j] = 0.0f;
    }
}

// ---------------------------------------------------------------------------
// 0b) Z (f32) -> bf16, 4 elements/thread
// ---------------------------------------------------------------------------
__global__ void zcvt_kernel(const float* __restrict__ Z,
                            unsigned* __restrict__ Zb32, long long n4) {
    long long i = (long long)blockIdx.x * blockDim.x + threadIdx.x;
    long long stride = (long long)gridDim.x * blockDim.x;
    for (long long j = i; j < n4; j += stride) {
        float4 f = ((const float4*)Z)[j];
        unsigned u0 = (unsigned)f2bfu(f.x) | ((unsigned)f2bfu(f.y) << 16);
        unsigned u1 = (unsigned)f2bfu(f.z) | ((unsigned)f2bfu(f.w) << 16);
        Zb32[2 * j]     = u0;
        Zb32[2 * j + 1] = u1;
    }
}

// ---------------------------------------------------------------------------
// 0c) pack a KxN f32 weight matrix into bf16 WMMA B-operand tiles:
//     P[((ct*(K/32) + ks)*32 + lane)*16 + e], e=2j+p ->
//       kk = ((j<4)?2j:16+2(j-4)) + (lane>>4)*8 + p, k = ks*32+kk,
//       col = ct*16 + (lane&15)
// ---------------------------------------------------------------------------
__global__ void packB_kernel(const float* __restrict__ W,
                             unsigned short* __restrict__ P, int K, int N) {
    int idx = blockIdx.x * blockDim.x + threadIdx.x;
    if (idx >= K * N) return;
    int e    = idx & 15;
    int lane = (idx >> 4) & 31;
    int tile = idx >> 9;
    int nks  = K / 32;
    int ks   = tile % nks;
    int ct   = tile / nks;
    int j = e >> 1, p = e & 1;
    int kk  = ((j < 4) ? 2 * j : 16 + 2 * (j - 4)) + (lane >> 4) * 8 + p;
    int k   = ks * 32 + kk;
    int col = ct * 16 + (lane & 15);
    P[idx] = f2bfu(W[(size_t)k * N + col]);
}

// ---------------------------------------------------------------------------
// 1) projections. grid=(ceil(E/16), 4 heads), block=512 (16 waves)
//    wave w: m = w>>2 in {q,k,v,g}, ct = w&3; 4 WMMAs over K=128.
// ---------------------------------------------------------------------------
__global__ __launch_bounds__(512)
void proj_kernel(const unsigned short* __restrict__ Zb,
                 const unsigned short* __restrict__ Wp,   // packed: (m*4+h)*8192
                 const float* __restrict__ Z,             // f32, for bproj
                 const float* __restrict__ Wb, const float* __restrict__ bg,
                 unsigned short* __restrict__ q16, unsigned short* __restrict__ k16,
                 unsigned short* __restrict__ v16, unsigned short* __restrict__ g16,
                 float* __restrict__ bproj, int E) {
    const int h    = blockIdx.y;
    const int r0   = blockIdx.x * 16;
    const int tid  = threadIdx.x;
    const int lane = tid & 31;
    const int w    = tid >> 5;   // 0..15
    const int m    = w >> 2;     // 0:q 1:k 2:v 3:g
    const int ct   = w & 3;

    const int rowA = lane & 15;
    const int hiK  = (lane >> 4) * 8;               // 0 or 8
    const int gr   = min(r0 + rowA, E - 1);         // clamp: loads always valid
    const unsigned short* Arow = Zb + (size_t)gr * 128 + hiK;
    const unsigned short* Bp   = Wp + ((size_t)(m * 4 + h)) * 8192
                                    + (size_t)ct * 4 * 512 + (size_t)lane * 16;

    v8f acc = {};
#pragma unroll
    for (int ks = 0; ks < 4; ++ks) {
        v16bf a = load_op(Arow + ks * 32, Arow + ks * 32 + 16);
        v16bf b = load_op(Bp + ks * 512, Bp + ks * 512 + 8);
        acc = __builtin_amdgcn_wmma_f32_16x16x32_bf16(false, a, false, b,
                                                      (short)0, acc, false, false);
    }

    // D tile: VGPR r -> row (r + 8*(lane>=16)), col = ct*16 + (lane&15)
    const int nCol  = ct * 16 + (lane & 15);
    const int mBase = (lane >> 4) * 8;
    unsigned short* dst = (m == 0 ? q16 : m == 1 ? k16 : m == 2 ? v16 : g16);
#pragma unroll
    for (int r = 0; r < 8; ++r) {
        const int row = r0 + mBase + r;
        if (row < E) {
            float x = acc[r];
            if (m == 3) x = 1.0f / (1.0f + __expf(-(x + bg[h * 64 + nCol])));  // gate
            dst[((size_t)h * E + row) * 64 + nCol] = f2bfu(x);
        }
    }

    // rank-1 b projection: 16 threads, one row each, float4 dots
    if (tid < 16) {
        const int row = r0 + tid;
        if (row < E) {
            const float4* zr  = (const float4*)(Z + (size_t)row * 128);
            const float4* wb4 = (const float4*)(Wb + (size_t)h * 128);
            float s = 0.f;
#pragma unroll 4
            for (int i = 0; i < 32; ++i) {
                float4 za = zr[i], wa = wb4[i];
                s += za.x * wa.x + za.y * wa.y + za.z * wa.z + za.w * wa.w;
            }
            bproj[(size_t)h * E + row] = s;
        }
    }
}

// ---------------------------------------------------------------------------
// 2) logits per (triangle, head) + segment max via atomicMax(enc)
// ---------------------------------------------------------------------------
__global__ void logits_kernel(const unsigned short* __restrict__ q16,
                              const unsigned short* __restrict__ k16,
                              const float* __restrict__ bproj,
                              const int* __restrict__ tc, const int* __restrict__ tik,
                              const int* __restrict__ tjk,
                              float* __restrict__ logits, unsigned* __restrict__ menc,
                              int T, int E) {
    const int gid = blockIdx.x * blockDim.x + threadIdx.x;
    if (gid >= T * 4) return;
    const int t = gid >> 2, h = gid & 3;
    const int c = tc[t], ik = tik[t], jk = tjk[t];
    const uint4* qp = (const uint4*)(q16 + ((size_t)h * E + c)  * 64);
    const uint4* kp = (const uint4*)(k16 + ((size_t)h * E + ik) * 64);
    float s = 0.f;
#pragma unroll
    for (int i = 0; i < 8; ++i) {
        uint4 qu = qp[i], ku = kp[i];
        s += lo16(qu.x) * lo16(ku.x) + hi16(qu.x) * hi16(ku.x);
        s += lo16(qu.y) * lo16(ku.y) + hi16(qu.y) * hi16(ku.y);
        s += lo16(qu.z) * lo16(ku.z) + hi16(qu.z) * hi16(ku.z);
        s += lo16(qu.w) * lo16(ku.w) + hi16(qu.w) * hi16(ku.w);
    }
    const float lg = s * INV_SQRT_H + bproj[(size_t)h * E + jk];
    logits[gid] = lg;
    atomicMax(&menc[(size_t)h * E + c], fenc(lg));
}

// ---------------------------------------------------------------------------
// 3) accumulate: a = exp(l - m); denom += a; S[c] += a * v[ik]
// ---------------------------------------------------------------------------
__global__ __launch_bounds__(256)
void accum_kernel(const float* __restrict__ logits, const unsigned* __restrict__ menc,
                  const unsigned short* __restrict__ v16,
                  const int* __restrict__ tc, const int* __restrict__ tik,
                  float* __restrict__ S, float* __restrict__ denom, int E) {
    const int t = blockIdx.x;
    __shared__ float aSh[4];
    __shared__ int cSh, ikSh;
    const int tid = threadIdx.x;
    if (tid == 0) { cSh = tc[t]; ikSh = tik[t]; }
    __syncthreads();
    const int c = cSh, ik = ikSh;
    if (tid < 4) {
        const float m = fdec(menc[(size_t)tid * E + c]);  // finite: segment non-empty
        const float a = __expf(logits[t * 4 + tid] - m);
        atomicAdd(&denom[(size_t)tid * E + c], a);
        aSh[tid] = a;
    }
    __syncthreads();
    const int h = tid >> 6, d = tid & 63;
    atomicAdd(&S[((size_t)h * E + c) * 64 + d],
              aSh[h] * bfu2f(v16[((size_t)h * E + ik) * 64 + d]));
}

// ---------------------------------------------------------------------------
// 4) finalize: val = S/denom (1 if count==0); feat(e, h*64+d) = g*val (bf16)
// ---------------------------------------------------------------------------
__global__ __launch_bounds__(256)
void finalize_kernel(const float* __restrict__ S, const float* __restrict__ denom,
                     const unsigned short* __restrict__ g16,
                     const int* __restrict__ count,
                     unsigned short* __restrict__ feat16, int E) {
    const int e = blockIdx.x;
    const int tid = threadIdx.x;
    const int h = tid >> 6, d = tid & 63;
    float val;
    if (count[e] == 0) {
        val = 1.0f;
    } else {
        const float den = denom[(size_t)h * E + e];
        val = S[((size_t)h * E + e) * 64 + d] / fmaxf(den, 1e-30f);
    }
    const float gt = bfu2f(g16[((size_t)h * E + e) * 64 + d]);
    feat16[(size_t)e * 256 + h * 64 + d] = f2bfu(gt * val);
}

// ---------------------------------------------------------------------------
// 5) out = feat(E,256) @ Wout(256,128) + bout. block=256 (8 waves), 8 WMMAs.
// ---------------------------------------------------------------------------
__global__ __launch_bounds__(256)
void outgemm_kernel(const unsigned short* __restrict__ feat16,
                    const unsigned short* __restrict__ WoutP,  // packed, nks=8
                    const float* __restrict__ bout,
                    float* __restrict__ out, int E) {
    const int r0   = blockIdx.x * 16;
    const int tid  = threadIdx.x;
    const int lane = tid & 31;
    const int ct   = tid >> 5;   // 0..7 -> col tile
    const int rowA = lane & 15;
    const int hiK  = (lane >> 4) * 8;
    const int gr   = min(r0 + rowA, E - 1);
    const unsigned short* Arow = feat16 + (size_t)gr * 256 + hiK;
    const unsigned short* Bp   = WoutP + (size_t)ct * 8 * 512 + (size_t)lane * 16;

    v8f acc = {};
#pragma unroll
    for (int ks = 0; ks < 8; ++ks) {
        v16bf a = load_op(Arow + ks * 32, Arow + ks * 32 + 16);
        v16bf b = load_op(Bp + ks * 512, Bp + ks * 512 + 8);
        acc = __builtin_amdgcn_wmma_f32_16x16x32_bf16(false, a, false, b,
                                                      (short)0, acc, false, false);
    }

    const int nCol  = ct * 16 + (lane & 15);
    const int mBase = (lane >> 4) * 8;
    const float bo  = bout[nCol];
#pragma unroll
    for (int r = 0; r < 8; ++r) {
        const int row = r0 + mBase + r;
        if (row < E) out[(size_t)row * 128 + nCol] = acc[r] + bo;
    }
}

// ---------------------------------------------------------------------------
extern "C" void kernel_launch(void* const* d_in, const int* in_sizes, int n_in,
                              void* d_out, int out_size, void* d_ws, size_t ws_size,
                              hipStream_t stream) {
    const float* Z    = (const float*)d_in[0];
    const float* Wq   = (const float*)d_in[1];
    const float* Wk   = (const float*)d_in[2];
    const float* Wv   = (const float*)d_in[3];
    const float* Wb   = (const float*)d_in[4];
    const float* Wg   = (const float*)d_in[5];
    const float* bg   = (const float*)d_in[6];
    const float* Wout = (const float*)d_in[7];
    const float* bout = (const float*)d_in[8];
    const int* tc     = (const int*)d_in[9];
    const int* tik    = (const int*)d_in[10];
    const int* tjk    = (const int*)d_in[11];
    const int* count  = (const int*)d_in[12];
    float* out        = (float*)d_out;

    const int E = in_sizes[0] / 128;
    const int T = in_sizes[9];

    size_t off = 0;
    auto alloc = [&](size_t bytes) {
        void* p = (char*)d_ws + off;
        off += (bytes + 255) & ~(size_t)255;
        return p;
    };
    const size_t HE = (size_t)4 * E;
    unsigned short* Zb     = (unsigned short*)alloc((size_t)E * 128 * 2);
    unsigned short* q16    = (unsigned short*)alloc(HE * 64 * 2);
    unsigned short* k16    = (unsigned short*)alloc(HE * 64 * 2);
    unsigned short* v16    = (unsigned short*)alloc(HE * 64 * 2);
    unsigned short* g16    = (unsigned short*)alloc(HE * 64 * 2);
    unsigned short* feat16 = (unsigned short*)alloc((size_t)E * 256 * 2);
    unsigned short* Wp     = (unsigned short*)alloc((size_t)16 * 8192 * 2);
    unsigned short* WoutP  = (unsigned short*)alloc((size_t)32768 * 2);
    float*    S      = (float*)alloc(HE * 64 * 4);
    float*    denom  = (float*)alloc(HE * 4);
    float*    bproj  = (float*)alloc(HE * 4);
    unsigned* menc   = (unsigned*)alloc(HE * 4);
    float*    logits = (float*)alloc((size_t)T * 4 * 4);

    const int rowTiles = (E + 15) / 16;

    init_kernel<<<2048, 256, 0, stream>>>(menc, denom, S, (long long)HE,
                                          (long long)HE * 64);
    zcvt_kernel<<<2048, 256, 0, stream>>>(Z, (unsigned*)Zb, (long long)E * 32);

    // pack 16 projection weight matrices (order m-major: q,k,v,g x heads)
    const float* Wmats[4] = {Wq, Wk, Wv, Wg};
    for (int m = 0; m < 4; ++m)
        for (int h = 0; h < 4; ++h)
            packB_kernel<<<(128 * 64 + 255) / 256, 256, 0, stream>>>(
                Wmats[m] + (size_t)h * 128 * 64, Wp + (size_t)(m * 4 + h) * 8192,
                128, 64);
    packB_kernel<<<(256 * 128 + 255) / 256, 256, 0, stream>>>(Wout, WoutP, 256, 128);

    proj_kernel<<<dim3(rowTiles, 4), 512, 0, stream>>>(Zb, Wp, Z, Wb, bg,
                                                       q16, k16, v16, g16, bproj, E);

    if (T > 0) {
        logits_kernel<<<(4 * T + 255) / 256, 256, 0, stream>>>(q16, k16, bproj,
                                                               tc, tik, tjk,
                                                               logits, menc, T, E);
        accum_kernel<<<T, 256, 0, stream>>>(logits, menc, v16, tc, tik, S, denom, E);
    }

    finalize_kernel<<<E, 256, 0, stream>>>(S, denom, g16, count, feat16, E);

    outgemm_kernel<<<rowTiles, 256, 0, stream>>>(feat16, WoutP, bout, out, E);
}
